// EncoderLayer_hybrid1_46651934769209
// MI455X (gfx1250) — matrix operand order repaired and checked by reading, not compiled, not run on hardware
//
#include <hip/hip_runtime.h>

typedef __attribute__((ext_vector_type(16))) _Float16 v16h;
typedef __attribute__((ext_vector_type(8)))  _Float16 v8h;
typedef __attribute__((ext_vector_type(8)))  float    v8f;

// Problem constants: B=8, T=2048, E=512, H=8, d=64
#define NB 8
#define NT 2048
#define NE 512
#define NH 8
#define ND 64
// log2(e)/8 : scores in log2 domain, s2 = -(q-k)^2 * C2
#define C2 (-0.18033688011112042f)
// LDS chunk row stride in bytes (64B payload + 16B pad, 16B aligned)
#define RSTRIDE 80
#define CHUNK_BYTES (64 * RSTRIDE)   // 5120 B per buffer

static __device__ __forceinline__ v8f wmma_f16(v16h a, v16h b, v8f c) {
    return __builtin_amdgcn_wmma_f32_16x16x32_f16(false, a, false, b, (short)0, c, false, false);
}

// CDNA5 async global->LDS copy (ASYNCcnt tracked), 16 bytes per lane.
// lds = wave-relative LDS byte address (low 32 bits of generic LDS pointer).
static __device__ __forceinline__ void async_ld16(unsigned lds, const void* g) {
    asm volatile("global_load_async_to_lds_b128 %0, %1, off"
                 :: "v"(lds), "v"(g) : "memory");
}
static __device__ __forceinline__ void wait_async_le1() {
    asm volatile("s_wait_asynccnt 0x1" ::: "memory");
}
static __device__ __forceinline__ void wait_async_le0() {
    asm volatile("s_wait_asynccnt 0x0" ::: "memory");
}
static __device__ __forceinline__ unsigned lds_off(const void* p) {
    return (unsigned)(unsigned long long)p;
}
// Load one 32B WMMA-B lane slice (two 16B LDS loads) from a staged chunk.
static __device__ __forceinline__ v16h ld_btile(const char* rowbase) {
    v8h lo = *(const v8h*)(rowbase);
    v8h hi = *(const v8h*)(rowbase + 32);
    return __builtin_shufflevector(lo, hi, 0, 1, 2, 3, 4, 5, 6, 7,
                                   8, 9, 10, 11, 12, 13, 14, 15);
}

// ---------------------------------------------------------------------------
// Kernel 0: Wmt[n][k] = (f16) Wm[k][n]   (512x512) — WMMA-B friendly layout
// ---------------------------------------------------------------------------
__global__ void __launch_bounds__(256) prep_wmt(const float* __restrict__ Wm,
                                                _Float16* __restrict__ Wmt) {
    int idx = blockIdx.x * 256 + threadIdx.x;     // 0 .. 262143
    int n = idx >> 9, k = idx & 511;
    Wmt[idx] = (_Float16)Wm[k * NE + n];
}

// ---------------------------------------------------------------------------
// Kernel 1: LayerNorm + Q/K scalars + V projection (one wave per token).
// Writes Qb[B*H, T], Kb[B*H, T] (f32) and Vt[B, H, 64, T] (f16, transposed
// so attention B-tiles are contiguous).
// ---------------------------------------------------------------------------
__global__ void __launch_bounds__(256) ln_qkv(const float* __restrict__ X,
                                              const float* __restrict__ Wq,
                                              const float* __restrict__ Wk,
                                              const float* __restrict__ Wv,
                                              float* __restrict__ Qb,
                                              float* __restrict__ Kb,
                                              _Float16* __restrict__ Vt) {
    __shared__ float    xnl[8][NE];   // 16 KB normalized x per token
    __shared__ _Float16 vsl[8][NE];   //  8 KB V staging for transpose
    const int tid  = threadIdx.x;
    const int wave = tid >> 5, lane = tid & 31;
    const int t  = blockIdx.x * 8 + wave;        // global token
    const int b  = t >> 11, tt = t & (NT - 1);

    // --- LayerNorm: each lane owns 16 contiguous floats ---
    const float4* xp = (const float4*)(X + (size_t)t * NE);
    float4 u[4];
    float s = 0.f, ss = 0.f;
#pragma unroll
    for (int i = 0; i < 4; ++i) {
        u[i] = xp[lane * 4 + i];
        s  += u[i].x + u[i].y + u[i].z + u[i].w;
        ss += u[i].x*u[i].x + u[i].y*u[i].y + u[i].z*u[i].z + u[i].w*u[i].w;
    }
#pragma unroll
    for (int off = 16; off >= 1; off >>= 1) {
        s  += __shfl_xor(s,  off, 32);
        ss += __shfl_xor(ss, off, 32);
    }
    const float mu   = s * (1.f / NE);
    const float var  = ss * (1.f / NE) - mu * mu;
    const float rstd = rsqrtf(var + 1e-5f);
#pragma unroll
    for (int i = 0; i < 4; ++i) {
        xnl[wave][lane * 16 + i * 4 + 0] = (u[i].x - mu) * rstd;
        xnl[wave][lane * 16 + i * 4 + 1] = (u[i].y - mu) * rstd;
        xnl[wave][lane * 16 + i * 4 + 2] = (u[i].z - mu) * rstd;
        xnl[wave][lane * 16 + i * 4 + 3] = (u[i].w - mu) * rstd;
    }
    __syncthreads();

    // --- Q/K scalars: lanes 0..7 -> Q heads, 8..15 -> K heads ---
    if (lane < 16) {
        const int h = lane & 7;
        const float* w  = (lane < 8 ? Wq : Wk) + h * ND;
        const float* xh = &xnl[wave][h * ND];
        float acc = 0.f;
#pragma unroll 8
        for (int i = 0; i < ND; ++i) acc += xh[i] * w[i];
        float* dst = (lane < 8 ? Qb : Kb);
        dst[(size_t)(b * NH + h) * NT + tt] = acc;
    }

    // --- V projection: 512 outputs / 32 lanes = 16 dots of 64 per lane ---
    for (int i = 0; i < 16; ++i) {
        const int o = lane + 32 * i;
        const int h = o >> 6, e = o & 63;
        const float* xh = &xnl[wave][h * ND];
        const float* w  = Wv + h * ND * ND + e;     // stride-64 column of Wv[h]
        float acc = 0.f;
#pragma unroll 8
        for (int dd = 0; dd < ND; ++dd) acc += xh[dd] * w[dd * ND];
        vsl[wave][o] = (_Float16)acc;
    }
    __syncthreads();

    // --- transpose write: Vt[(b*512 + h*64+e)*T + t], 8 tokens per block ---
    const int tt0 = (blockIdx.x & 255) * 8;
    for (int rr = tid; rr < NE; rr += 256) {
        v8h pk;
#pragma unroll
        for (int tok = 0; tok < 8; ++tok) pk[tok] = vsl[tok][rr];
        *(v8h*)(Vt + ((size_t)(b * NE + rr)) * NT + tt0) = pk;
    }
}

// ---------------------------------------------------------------------------
// Kernel 2: distance attention, flash-style with analytic scores.
// Block = 8 waves sharing one (b,h); wave = one 16-query tile producing a
// 16x64 O tile via 4x v_wmma_f32_16x16x32_f16 per 32-key chunk.
// V chunks (4 KB, identical for all 8 waves) are double-buffered in LDS via
// async global->LDS copies (ASYNCcnt).
// ---------------------------------------------------------------------------
__global__ void __launch_bounds__(256) attn(const float* __restrict__ Qb,
                                            const float* __restrict__ Kb,
                                            const _Float16* __restrict__ Vt,
                                            float* __restrict__ HR) {
    __shared__ __align__(16) float Kl[NT];            // 8 KB keys for this (b,h)
    __shared__ __align__(16) char  Vl[2][CHUNK_BYTES]; // 10 KB double buffer
    const int tid  = threadIdx.x;
    const int wave = tid >> 5, lane = tid & 31;
    const int half = lane >> 4, r = lane & 15;
    const int bh = blockIdx.x >> 4;               // 0..63
    const int qt = (blockIdx.x & 15) * 8 + wave;  // 0..127 query tile

    // stage K in LDS (2048 floats, 256 threads x 2 float4)
    const float4* Kg  = (const float4*)(Kb + (size_t)bh * NT);
    float4* Kl4 = (float4*)Kl;
    Kl4[tid]       = Kg[tid];
    Kl4[tid + 256] = Kg[tid + 256];

    // kick off async copy of V chunk 0 into buffer 0 (hidden behind pass 1)
    const char* gV   = (const char*)(Vt + (size_t)bh * ND * NT);
    const int   crow = tid >> 2;              // 0..63 (d-row of chunk)
    const int   cpart = (tid & 3) * 16;       // 16B slice within 64B row
    async_ld16(lds_off(&Vl[0][crow * RSTRIDE + cpart]),
               gV + (size_t)crow * (NT * 2) + cpart);

    __syncthreads();

    const float q = Qb[(size_t)bh * NT + qt * 16 + r];

    // --- pass 1: row max (log2 domain), lane pairs split keys 4-at-a-time ---
    float m2 = -3.0e38f;
#pragma unroll 4
    for (int i = 0; i < 256; ++i) {
        const float4 kv = Kl4[i * 2 + half];
        const float d0 = q - kv.x, d1 = q - kv.y, d2 = q - kv.z, d3 = q - kv.w;
        m2 = fmaxf(m2, fmaxf(fmaxf(C2 * d0 * d0, C2 * d1 * d1),
                             fmaxf(C2 * d2 * d2, C2 * d3 * d3)));
    }
    m2 = fmaxf(m2, __shfl_xor(m2, 16, 32));
    // --- pass 1b: row sum of exp2 ---
    float l = 0.f;
#pragma unroll 4
    for (int i = 0; i < 256; ++i) {
        const float4 kv = Kl4[i * 2 + half];
        const float d0 = q - kv.x, d1 = q - kv.y, d2 = q - kv.z, d3 = q - kv.w;
        l += __builtin_amdgcn_exp2f(C2 * d0 * d0 - m2) +
             __builtin_amdgcn_exp2f(C2 * d1 * d1 - m2) +
             __builtin_amdgcn_exp2f(C2 * d2 * d2 - m2) +
             __builtin_amdgcn_exp2f(C2 * d3 * d3 - m2);
    }
    l += __shfl_xor(l, 16, 32);
    const float invl = __builtin_amdgcn_rcpf(l);

    // --- pass 2: P tile (f16 A) x V (f16 B from LDS) -> 16x64 f32 acc ---
    v8f c0 = {}, c1 = {}, c2 = {}, c3 = {};
    for (int jc = 0; jc < 64; ++jc) {
        if (jc + 1 < 64) {   // prefetch next chunk into the other buffer
            async_ld16(lds_off(&Vl[(jc + 1) & 1][crow * RSTRIDE + cpart]),
                       gV + (size_t)crow * (NT * 2) + (jc + 1) * 64 + cpart);
            wait_async_le1();
        } else {
            wait_async_le0();
        }
        __syncthreads();     // chunk jc visible to all waves

        const int j0 = jc * 32;
        v16h a;
#pragma unroll
        for (int e = 0; e < 16; ++e) {
            const int ko = (e < 8 ? e : e + 8) + half * 8;  // ISA 16-bit A layout
            const float dq = q - Kl[j0 + ko];
            const float p  = __builtin_amdgcn_exp2f(C2 * dq * dq - m2) * invl;
            a[e] = (_Float16)p;
        }
        const char* vb = &Vl[jc & 1][(size_t)r * RSTRIDE + half * 16];
        const v16h b0 = ld_btile(vb);
        const v16h b1 = ld_btile(vb + 16 * RSTRIDE);
        const v16h b2 = ld_btile(vb + 32 * RSTRIDE);
        const v16h b3 = ld_btile(vb + 48 * RSTRIDE);
        c0 = wmma_f16(a, b0, c0);
        c1 = wmma_f16(a, b1, c1);
        c2 = wmma_f16(a, b2, c2);
        c3 = wmma_f16(a, b3, c3);

        __syncthreads();     // all waves done reading buf jc&1 before reuse
    }

    // --- writeout: HR[b, t, h*64 + n] (f32), C layout: row v+8*half, col r ---
    const int bb = bh >> 3, hh = bh & 7;
    float* Ho = HR + ((size_t)(bb * NT + qt * 16)) * NE + hh * ND + r;
#pragma unroll
    for (int v = 0; v < 8; ++v) {
        float* p = Ho + (size_t)(v + half * 8) * NE;
        p[0]  = c0[v];
        p[16] = c1[v];
        p[32] = c2[v];
        p[48] = c3[v];
    }
}

// ---------------------------------------------------------------------------
// Kernel 3: out = HR @ Wm + input1.  [16384,512]x[512,512].
// 8 waves per block share one 64-column group (same Wmt tiles): Wmt chunks
// double-buffered in LDS via async copies; each wave owns a 16-token row tile.
// ---------------------------------------------------------------------------
__global__ void __launch_bounds__(256) merge(const float* __restrict__ HR,
                                             const _Float16* __restrict__ Wmt,
                                             const float* __restrict__ X,
                                             float* __restrict__ Out) {
    __shared__ __align__(16) char Wl[2][CHUNK_BYTES];  // 10 KB double buffer
    const int tid  = threadIdx.x;
    const int wave = tid >> 5, lane = tid & 31;
    const int half = lane >> 4, r = lane & 15;
    const int nt = blockIdx.x >> 7;                   // 0..7  column group
    const int mt = (blockIdx.x & 127) * 8 + wave;     // 0..1023 row tile

    // async prologue: Wmt chunk 0 -> buffer 0
    const char* gW   = (const char*)(Wmt + (size_t)nt * 64 * NE);
    const int   crow = tid >> 2;
    const int   cpart = (tid & 3) * 16;
    async_ld16(lds_off(&Wl[0][crow * RSTRIDE + cpart]),
               gW + (size_t)crow * (NE * 2) + cpart);

    const float* Ap = HR + ((size_t)(mt * 16 + r)) * NE;
    v8f c0 = {}, c1 = {}, c2 = {}, c3 = {};
    for (int kc = 0; kc < 16; ++kc) {
        if (kc + 1 < 16) {
            async_ld16(lds_off(&Wl[(kc + 1) & 1][crow * RSTRIDE + cpart]),
                       gW + (size_t)crow * (NE * 2) + (kc + 1) * 64 + cpart);
            wait_async_le1();
        } else {
            wait_async_le0();
        }
        __syncthreads();

        const int k0 = kc * 32;
        const float4* a0 = (const float4*)(Ap + k0 + half * 8);
        const float4* a1 = (const float4*)(Ap + k0 + 16 + half * 8);
        const float4 f0 = a0[0], f1 = a0[1], f2 = a1[0], f3 = a1[1];
        v16h a;
        a[0]  = (_Float16)f0.x; a[1]  = (_Float16)f0.y;
        a[2]  = (_Float16)f0.z; a[3]  = (_Float16)f0.w;
        a[4]  = (_Float16)f1.x; a[5]  = (_Float16)f1.y;
        a[6]  = (_Float16)f1.z; a[7]  = (_Float16)f1.w;
        a[8]  = (_Float16)f2.x; a[9]  = (_Float16)f2.y;
        a[10] = (_Float16)f2.z; a[11] = (_Float16)f2.w;
        a[12] = (_Float16)f3.x; a[13] = (_Float16)f3.y;
        a[14] = (_Float16)f3.z; a[15] = (_Float16)f3.w;
        const char* wb = &Wl[kc & 1][(size_t)r * RSTRIDE + half * 16];
        const v16h b0 = ld_btile(wb);
        const v16h b1 = ld_btile(wb + 16 * RSTRIDE);
        const v16h b2 = ld_btile(wb + 32 * RSTRIDE);
        const v16h b3 = ld_btile(wb + 48 * RSTRIDE);
        c0 = wmma_f16(a, b0, c0);
        c1 = wmma_f16(a, b1, c1);
        c2 = wmma_f16(a, b2, c2);
        c3 = wmma_f16(a, b3, c3);

        __syncthreads();
    }
    // residual + store
    const size_t obase = ((size_t)(mt * 16)) * NE + nt * 64 + r;
#pragma unroll
    for (int v = 0; v < 8; ++v) {
        const size_t o = obase + (size_t)(v + half * 8) * NE;
        Out[o]      = c0[v] + X[o];
        Out[o + 16] = c1[v] + X[o + 16];
        Out[o + 32] = c2[v] + X[o + 32];
        Out[o + 48] = c3[v] + X[o + 48];
    }
}

// ---------------------------------------------------------------------------
extern "C" void kernel_launch(void* const* d_in, const int* in_sizes, int n_in,
                              void* d_out, int out_size, void* d_ws, size_t ws_size,
                              hipStream_t stream) {
    (void)in_sizes; (void)n_in; (void)out_size; (void)ws_size;
    const float* X  = (const float*)d_in[0];   // [8,2048,512]
    const float* Wq = (const float*)d_in[1];   // [8,64]
    const float* Wk = (const float*)d_in[2];   // [8,64]
    const float* Wv = (const float*)d_in[3];   // [8,64,64]
    const float* Wm = (const float*)d_in[4];   // [512,512]
    float* out = (float*)d_out;

    char* ws = (char*)d_ws;
    float*    Qb  = (float*)ws;                                    // 512 KB
    float*    Kb  = (float*)(ws + (512u << 10));                   // 512 KB
    _Float16* Vt  = (_Float16*)(ws + (1024u << 10));               //  16 MB
    _Float16* Wmt = (_Float16*)(ws + (1024u << 10) + (16u << 20)); // 512 KB
    float*    HR  = (float*)(ws + (1536u << 10) + (16u << 20));    //  32 MB

    prep_wmt<<<1024, 256, 0, stream>>>(Wm, Wmt);
    ln_qkv  <<<2048, 256, 0, stream>>>(X, Wq, Wk, Wv, Qb, Kb, Vt);
    attn    <<<1024, 256, 0, stream>>>(Qb, Kb, Vt, HR);
    merge   <<<1024, 256, 0, stream>>>(HR, Wmt, X, out);
}
